// ChildSumTreeLSTMVerbose_43911745635052
// MI455X (gfx1250) — compile-verified, matrix operand order checked by reading, not run on hardware
//
#include <hip/hip_runtime.h>
#include <hip/hip_bf16.h>
#include <math.h>

typedef __attribute__((ext_vector_type(16))) _Float16 v16h;
typedef __attribute__((ext_vector_type(8)))  float    v8f;

#define NN   65536
#define DD   300
#define HH   256
#define KSX  10      // ceil(300/32) -> K padded to 320
#define KSH  8       // 256/32
#define TILE_HALFS  (4 * KSH * 32 * 16)     // 16384 halfs = 32 KiB per 16-node tile

__device__ __forceinline__ float sigmoidf_(float x) {
    return 1.0f / (1.0f + __expf(-x));
}

// Fragment map (16-bit A/B operands, wave32):
//   element (m, k) -> lane = (m&15) + 16*((k>>3)&1), half = (k&7) + 8*(k>>4)
// ---------------------------------------------------------------------------
__global__ void k_swizzle_in(const float* __restrict__ in, _Float16* __restrict__ sw) {
    int id = blockIdx.x * 256 + threadIdx.x;
    const int total = (NN / 16) * KSX * 32 * 16;
    if (id >= total) return;
    int hf   = id & 15;
    int lane = (id >> 4) & 31;
    int ks   = (id >> 9) % KSX;
    int rt   = id / (KSX * 512);
    int kk   = (hf & 7) + 8 * (lane >> 4) + 16 * (hf >> 3);
    int k    = ks * 32 + kk;
    int row  = rt * 16 + (lane & 15);
    float v  = (k < DD) ? in[row * DD + k] : 0.0f;
    sw[id] = (_Float16)v;
}

__global__ void k_swizzle_w(const float* __restrict__ w, _Float16* __restrict__ sw,
                            int Dw, int KS) {
    int id = blockIdx.x * 256 + threadIdx.x;
    int total = (HH / 16) * KS * 512;
    if (id >= total) return;
    int hf   = id & 15;
    int lane = (id >> 4) & 31;
    int ks   = (id >> 9) % KS;
    int ct   = id / (KS * 512);
    int kk   = (hf & 7) + 8 * (lane >> 4) + 16 * (hf >> 3);
    int k    = ks * 32 + kk;
    int n    = ct * 16 + (lane & 15);
    float v  = (k < Dw) ? w[n * Dw + k] : 0.0f;
    sw[id] = (_Float16)v;
}

// ---------------------------------------------------------------------------
// Fused input projections: one A-fragment load feeds 3 WMMAs (ix, ux, fi).
// grid = (NN/16, 2), block = 256 (8 waves); wave -> one 16-col tile.
// reuse_a hint set on the 2nd/3rd WMMA (identical instruction, same A regs).
// ---------------------------------------------------------------------------
__global__ void k_proj(const _Float16* __restrict__ swIn,
                       const _Float16* __restrict__ swIX, const _Float16* __restrict__ swUX,
                       const _Float16* __restrict__ swFI,
                       const float* __restrict__ bix, const float* __restrict__ bux,
                       const float* __restrict__ bfi,
                       float* __restrict__ yix, float* __restrict__ yux,
                       float* __restrict__ yfi) {
    const int rt   = blockIdx.x;
    const int lane = threadIdx.x & 31;
    const int wave = threadIdx.x >> 5;
    const int ct   = blockIdx.y * 8 + wave;

    v8f aci = {}, acu = {}, acf = {};
    #pragma unroll
    for (int ks = 0; ks < KSX; ++ks) {
        v16h a  = *(const v16h*)(swIn + ((size_t)(rt * KSX + ks) * 32 + lane) * 16);
        v16h bi = *(const v16h*)(swIX + ((size_t)(ct * KSX + ks) * 32 + lane) * 16);
        v16h bu = *(const v16h*)(swUX + ((size_t)(ct * KSX + ks) * 32 + lane) * 16);
        v16h bf = *(const v16h*)(swFI + ((size_t)(ct * KSX + ks) * 32 + lane) * 16);
        if (ks + 1 < KSX)
            __builtin_prefetch(swIn + ((size_t)(rt * KSX + ks + 1) * 32 + lane) * 16, 0, 0);
        aci = __builtin_amdgcn_wmma_f32_16x16x32_f16(false, a, false, bi, (short)0, aci, false, false);
        acu = __builtin_amdgcn_wmma_f32_16x16x32_f16(false, a, false, bu, (short)0, acu, true,  false);
        acf = __builtin_amdgcn_wmma_f32_16x16x32_f16(false, a, false, bf, (short)0, acf, true,  false);
    }
    const int n  = ct * 16 + (lane & 15);
    const int mb = (lane >> 4) * 8;
    const float bvi = bix[n], bvu = bux[n], bvf = bfi[n];
    #pragma unroll
    for (int r = 0; r < 8; ++r) {
        size_t off = (size_t)(rt * 16 + mb + r) * HH + n;
        yix[off] = aci[r] + bvi;
        yux[off] = acu[r] + bvu;
        yfi[off] = acf[r] + bvf;
    }
}

// ---------------------------------------------------------------------------
// Scatter h (f16) into the parent level's pre-swizzled staging region so the
// parent kernel can DMA it straight into LDS. Same fragment map as above.
// ---------------------------------------------------------------------------
__device__ __forceinline__ void scatter_h(_Float16* __restrict__ wr, int node, int n,
                                          float hn, int s_par) {
    int p  = (node - 1) >> 2;
    int j  = (node - 1) & 3;
    int m2 = p - s_par;
    int tile = m2 >> 4, mm = m2 & 15;
    int ks = n >> 5, kk = n & 31;
    int lane2 = mm + 16 * ((kk >> 3) & 1);
    int hf    = (kk & 7) + 8 * (kk >> 4);
    wr[(size_t)tile * TILE_HALFS + (size_t)((j * KSH + ks) * 32 + lane2) * 16 + hf] =
        (_Float16)hn;
}

// Leaf level: c = sigmoid(ix+ih_b)*tanh(ux+uh_b), h = tanh(c); h goes straight
// into the level-7 staging region (f16, fragment layout).
__global__ void k_leaf(const float* __restrict__ ix, const float* __restrict__ ux,
                       const float* __restrict__ ihb, const float* __restrict__ uhb,
                       float* __restrict__ c, _Float16* __restrict__ wr,
                       int s, int e, int s_par) {
    int id = blockIdx.x * 256 + threadIdx.x;
    int total = (e - s) * HH;
    if (id >= total) return;
    int node = s + (id >> 8);
    int n    = id & 255;
    size_t off = (size_t)node * HH + n;
    float ig = sigmoidf_(ix[off] + ihb[n]);
    float ug = tanhf(ux[off] + uhb[n]);
    float cv = ig * ug;
    c[off] = cv;
    scatter_h(wr, node, n, tanhf(cv), s_par);
}

// ---------------------------------------------------------------------------
// Interior level: block = 16-node tile x H=256, 512 threads (16 waves).
// Children-h A-fragments arrive pre-swizzled in `rd`; gather is a bulk
// GLOBAL_LOAD_ASYNC_TO_LDS_B128 DMA (32 KiB contiguous), then 8 K-steps of
// 6 WMMAs per wave (i, u, 4 per-child forget gates).
// ---------------------------------------------------------------------------
__global__ void k_level(float* __restrict__ c,
                        const float* __restrict__ ix, const float* __restrict__ ux,
                        const float* __restrict__ fi,
                        const _Float16* __restrict__ swIH,
                        const _Float16* __restrict__ swUH,
                        const _Float16* __restrict__ swFH,
                        const float* __restrict__ ihb, const float* __restrict__ uhb,
                        const float* __restrict__ fhb,
                        const _Float16* __restrict__ rd, _Float16* __restrict__ wr,
                        float* __restrict__ hroot,
                        int s, int e, int s_par) {
    __shared__ _Float16 ldsA[TILE_HALFS];   // 32 KiB

    const int tid = threadIdx.x;
    const int bx  = blockIdx.x;

    // ---- async DMA: 32 KiB staging block -> LDS (2048 x b128) ----
    {
        unsigned long long ga0 = (unsigned long long)(const void*)(rd) +
                                 (unsigned long long)bx * (TILE_HALFS * 2);
        unsigned ldsBase = (unsigned)(size_t)(&ldsA[0]);
        #pragma unroll
        for (int t = 0; t < 4; ++t) {
            int i = tid + t * 512;
            unsigned lo = ldsBase + (unsigned)i * 16u;
            unsigned long long ga = ga0 + (unsigned long long)i * 16ull;
            asm volatile("global_load_async_to_lds_b128 %0, %1, off"
                         :: "v"(lo), "v"(ga) : "memory");
        }
        asm volatile("s_wait_asynccnt 0x0" ::: "memory");
    }
    __syncthreads();

    const int lane = tid & 31;
    const int wave = tid >> 5;          // 0..15 -> 16-col tile
    const int ct   = wave;

    v8f ai = {}, au = {}, af0 = {}, af1 = {}, af2 = {}, af3 = {};
    #pragma unroll
    for (int ks = 0; ks < KSH; ++ks) {
        v16h a0 = *(const v16h*)&ldsA[((0 * KSH + ks) * 32 + lane) * 16];
        v16h a1 = *(const v16h*)&ldsA[((1 * KSH + ks) * 32 + lane) * 16];
        v16h a2 = *(const v16h*)&ldsA[((2 * KSH + ks) * 32 + lane) * 16];
        v16h a3 = *(const v16h*)&ldsA[((3 * KSH + ks) * 32 + lane) * 16];
        v16h asum = (a0 + a1) + (a2 + a3);          // packed f16 adds
        v16h bi = *(const v16h*)(swIH + ((size_t)(ct * KSH + ks) * 32 + lane) * 16);
        v16h bu = *(const v16h*)(swUH + ((size_t)(ct * KSH + ks) * 32 + lane) * 16);
        v16h bf = *(const v16h*)(swFH + ((size_t)(ct * KSH + ks) * 32 + lane) * 16);
        ai  = __builtin_amdgcn_wmma_f32_16x16x32_f16(false, asum, false, bi, (short)0, ai,  false, false);
        au  = __builtin_amdgcn_wmma_f32_16x16x32_f16(false, asum, false, bu, (short)0, au,  true,  false);
        af0 = __builtin_amdgcn_wmma_f32_16x16x32_f16(false, a0,   false, bf, (short)0, af0, false, false);
        af1 = __builtin_amdgcn_wmma_f32_16x16x32_f16(false, a1,   false, bf, (short)0, af1, false, true);
        af2 = __builtin_amdgcn_wmma_f32_16x16x32_f16(false, a2,   false, bf, (short)0, af2, false, true);
        af3 = __builtin_amdgcn_wmma_f32_16x16x32_f16(false, a3,   false, bf, (short)0, af3, false, true);
    }

    // ---- epilogue: gates + cell update ----
    const int n  = ct * 16 + (lane & 15);
    const int mb = (lane >> 4) * 8;
    const float ihbv = ihb[n], uhbv = uhb[n], fhbv = fhb[n];
    #pragma unroll
    for (int r = 0; r < 8; ++r) {
        int node = s + bx * 16 + mb + r;
        if (node < e) {
            size_t off = (size_t)node * HH + n;
            float ig = sigmoidf_(ai[r] + ix[off] + ihbv);
            float ug = tanhf(au[r] + ux[off] + uhbv);
            float fbase = fi[off] + fhbv;
            float afv[4] = { af0[r], af1[r], af2[r], af3[r] };
            float fc = 0.0f;
            #pragma unroll
            for (int j = 0; j < 4; ++j) {
                int cidx = 4 * node + 1 + j;
                if (cidx < NN) {
                    float fg = sigmoidf_(afv[j] + fbase);
                    fc += fg * c[(size_t)cidx * HH + n];
                }
            }
            float cn = ig * ug + fc;
            c[off] = cn;
            float hn = tanhf(cn);
            if (wr) scatter_h(wr, node, n, hn, s_par);
            else    hroot[n] = hn;     // only node 0 reaches here at the root level
        }
    }
}

__global__ void k_root_out(const float* __restrict__ hroot, const float* __restrict__ c,
                           float* __restrict__ out) {
    int i = threadIdx.x;
    if (i < 256)       out[i] = hroot[i];
    else if (i < 512)  out[i] = c[i - 256];
}

// ---------------------------------------------------------------------------
extern "C" void kernel_launch(void* const* d_in, const int* in_sizes, int n_in,
                              void* d_out, int out_size, void* d_ws, size_t ws_size,
                              hipStream_t stream) {
    const float* inputs = (const float*)d_in[0];
    const float* ix_w = (const float*)d_in[1];  const float* ix_b = (const float*)d_in[2];
    const float* ih_w = (const float*)d_in[3];  const float* ih_b = (const float*)d_in[4];
    const float* ux_w = (const float*)d_in[5];  const float* ux_b = (const float*)d_in[6];
    const float* uh_w = (const float*)d_in[7];  const float* uh_b = (const float*)d_in[8];
    const float* fi_w = (const float*)d_in[9];  const float* fi_b = (const float*)d_in[10];
    const float* fh_w = (const float*)d_in[11]; const float* fh_b = (const float*)d_in[12];

    static const int bounds[10] = {0, 1, 5, 21, 85, 341, 1365, 5461, 21845, 65536};

    // ---- workspace carve-up ----
    char* p = (char*)d_ws;
    const size_t szSwIn = (size_t)NN * (KSX * 32) * sizeof(_Float16);
    const size_t szSwWx = (size_t)(HH / 16) * KSX * 512 * sizeof(_Float16);
    const size_t szSwWh = (size_t)(HH / 16) * KSH * 512 * sizeof(_Float16);
    const size_t szProj = (size_t)NN * HH * sizeof(float);

    _Float16* swIn = (_Float16*)p;  p += szSwIn;
    _Float16* swIX = (_Float16*)p;  p += szSwWx;
    _Float16* swUX = (_Float16*)p;  p += szSwWx;
    _Float16* swFI = (_Float16*)p;  p += szSwWx;
    _Float16* swIH = (_Float16*)p;  p += szSwWh;
    _Float16* swUH = (_Float16*)p;  p += szSwWh;
    _Float16* swFH = (_Float16*)p;  p += szSwWh;
    float* ix = (float*)p;          p += szProj;
    float* ux = (float*)p;          p += szProj;
    float* fi = (float*)p;          p += szProj;
    float* c  = (float*)p;          p += szProj;
    float* hroot = (float*)p;       p += 1024;

    // per-level staging regions (children-h in fragment layout, 32 KiB / tile)
    _Float16* region[8];
    size_t regionBytes[8];
    for (int lv = 0; lv < 8; ++lv) {
        int tiles = (bounds[lv + 1] - bounds[lv] + 15) / 16;
        region[lv] = (_Float16*)p;
        regionBytes[lv] = (size_t)tiles * TILE_HALFS * sizeof(_Float16);
        p += regionBytes[lv];
    }

    // 1) swizzle activations + weights into WMMA fragment layouts
    {
        int total = (NN / 16) * KSX * 512;
        k_swizzle_in<<<(total + 255) / 256, 256, 0, stream>>>(inputs, swIn);
        int totx = (HH / 16) * KSX * 512;
        int toth = (HH / 16) * KSH * 512;
        k_swizzle_w<<<(totx + 255) / 256, 256, 0, stream>>>(ix_w, swIX, DD, KSX);
        k_swizzle_w<<<(totx + 255) / 256, 256, 0, stream>>>(ux_w, swUX, DD, KSX);
        k_swizzle_w<<<(totx + 255) / 256, 256, 0, stream>>>(fi_w, swFI, DD, KSX);
        k_swizzle_w<<<(toth + 255) / 256, 256, 0, stream>>>(ih_w, swIH, HH, KSH);
        k_swizzle_w<<<(toth + 255) / 256, 256, 0, stream>>>(uh_w, swUH, HH, KSH);
        k_swizzle_w<<<(toth + 255) / 256, 256, 0, stream>>>(fh_w, swFH, HH, KSH);
    }

    // 2) fused input projections (WMMA, one A load -> 3 WMMAs)
    {
        dim3 grid(NN / 16, 2);
        k_proj<<<grid, 256, 0, stream>>>(swIn, swIX, swUX, swFI,
                                         ix_b, ux_b, fi_b, ix, ux, fi);
    }

    // 3) zero level-7 staging region (covers masked/non-existent children)
    hipMemsetAsync(region[7], 0, regionBytes[7], stream);

    // 4) leaves (elementwise) write h straight into region[7]
    {
        int s = bounds[8], e = bounds[9];
        int total = (e - s) * HH;
        k_leaf<<<(total + 255) / 256, 256, 0, stream>>>(ix, ux, ih_b, uh_b, c,
                                                        region[7], s, e, bounds[7]);
    }

    // 5) interior levels 7..0 (stream order = dependency)
    for (int lv = 7; lv >= 0; --lv) {
        int s = bounds[lv], e = bounds[lv + 1];
        int tiles = (e - s + 15) / 16;
        _Float16* wr = (lv > 0) ? region[lv - 1] : (_Float16*)nullptr;
        int s_par = (lv > 0) ? bounds[lv - 1] : 0;
        k_level<<<tiles, 512, 0, stream>>>(c, ix, ux, fi,
                                           swIH, swUH, swFH,
                                           ih_b, uh_b, fh_b,
                                           region[lv], wr, hroot, s, e, s_par);
    }

    // 6) emit concat(h[0], c[0])
    k_root_out<<<1, 512, 0, stream>>>(hroot, c, (float*)d_out);
}
